// TransformerBlock_16750372454932
// MI455X (gfx1250) — compile-verified, matrix operand order checked by reading, not run on hardware
//
#include <hip/hip_runtime.h>
#include <cstdint>

// ---------------- CDNA5 types ----------------
typedef __attribute__((ext_vector_type(16))) __bf16        v16bf;
typedef __attribute__((ext_vector_type(8)))  float         v8f;
typedef __attribute__((ext_vector_type(4)))  unsigned int  u32x4;
typedef __attribute__((ext_vector_type(4)))  int           i32x4v;

#define AS_GLOBAL __attribute__((address_space(1)))
#define AS_LOCAL  __attribute__((address_space(3)))

// Async global->LDS copies (CDNA5 GLOBAL_LOAD_ASYNC_TO_LDS_*, ASYNCcnt).
#if defined(__has_builtin)
#  if __has_builtin(__builtin_amdgcn_global_load_async_to_lds_b128)
#    define HAVE_ASYNC_LDS 1
#  endif
#endif
#ifndef HAVE_ASYNC_LDS
#  define HAVE_ASYNC_LDS 0
#endif

#if HAVE_ASYNC_LDS
__device__ __forceinline__ void async_cp16(const unsigned short* gsrc,
                                           unsigned short* ldst) {
  unsigned short* g = const_cast<unsigned short*>(gsrc);
  __builtin_amdgcn_global_load_async_to_lds_b128(
      (AS_GLOBAL i32x4v*)g, (AS_LOCAL i32x4v*)ldst, 0, 0);
}
#  if __has_builtin(__builtin_amdgcn_s_wait_asynccnt)
#    define WAIT_ASYNC() __builtin_amdgcn_s_wait_asynccnt(0)
#  else
#    define WAIT_ASYNC() asm volatile("s_wait_asynccnt 0x0" ::: "memory")
#  endif
#endif

struct FragU { u32x4 lo, hi; };          // 32 bytes == 16 bf16
static_assert(sizeof(FragU) == 32, "frag size");

__device__ __forceinline__ unsigned short f2bf(float f) {
  unsigned u = __float_as_uint(f);
  unsigned r = u + 0x7fffu + ((u >> 16) & 1u);   // round-to-nearest-even
  return (unsigned short)(r >> 16);
}

// Load one 16x32 (A) / 32x16 (B, symmetric) bf16 WMMA fragment.
// Per CDNA5 ISA 16-bit A layout: lanes 0-15 hold K{0..7}U{16..23},
// lanes 16-31 hold K{8..15}U{24..31}, for row = given.
__device__ __forceinline__ v16bf load_frag(const unsigned short* base, int row,
                                           int rowStride, int k0, int lane) {
  const int khalf = (lane >> 4) * 8;
  const unsigned short* p = base + (size_t)row * rowStride + k0 + khalf;
  FragU f;
  f.lo = *(const u32x4*)(p);
  f.hi = *(const u32x4*)(p + 16);
  return __builtin_bit_cast(v16bf, f);
}

__device__ __forceinline__ v8f wmma_bf16(v16bf a, v16bf b, v8f c) {
  return __builtin_amdgcn_wmma_f32_16x16x32_bf16(false, a, false, b,
                                                 (short)0, c, false, false);
}

// ---------------- dims ----------------
constexpr int Bc  = 4;
constexpr int Sc  = 4096;
constexpr int Dc  = 1024;
constexpr int Hc  = 16;
constexpr int DHc = 64;
constexpr int FFc = 2048;
constexpr int Mrows = Bc * Sc;           // 16384

// ---------------- fp32 -> bf16 convert ----------------
__global__ void cvt_bf16(const float* __restrict__ in,
                         unsigned short* __restrict__ out, int n) {
  int i = blockIdx.x * blockDim.x + threadIdx.x;
  if (i < n) out[i] = f2bf(in[i]);
}

// ---------------- RMSNorm -> bf16 ----------------
__global__ __launch_bounds__(256)
void rmsnorm_bf16(const float* __restrict__ x, const float* __restrict__ scale,
                  unsigned short* __restrict__ out) {
  __shared__ float red[256];
  const int row = blockIdx.x;
  const int tid = threadIdx.x;
  const float* xr = x + (size_t)row * Dc;
  float v[4];
  float s = 0.f;
#pragma unroll
  for (int i = 0; i < 4; ++i) { v[i] = xr[tid + i * 256]; s += v[i] * v[i]; }
  red[tid] = s;
  __syncthreads();
  for (int off = 128; off > 0; off >>= 1) {
    if (tid < off) red[tid] += red[tid + off];
    __syncthreads();
  }
  const float inv = rsqrtf(red[0] * (1.0f / (float)Dc) + 1e-6f);
#pragma unroll
  for (int i = 0; i < 4; ++i) {
    const int c = tid + i * 256;
    out[(size_t)row * Dc + c] = f2bf(v[i] * inv * scale[c]);
  }
}

// ---------------- Generic bf16 WMMA GEMM ----------------
// out[M,N] = A[M,K] @ W[N,K]^T + bias (+res) (+act), store modes:
//  mode 0: linear (optional fp32 out, optional bf16 out)
//  mode 1: q/k scatter  -> outB[b,h,dh,s]  (f = dh*H + h, m = b*S + s)
//  mode 2: v scatter    -> outB[b,h,s,e]   (f = e*H + h)
__global__ __launch_bounds__(256)
void gemm_bf16(const unsigned short* __restrict__ A,
               const unsigned short* __restrict__ W,
               const float* __restrict__ bias,
               const float* __restrict__ res,
               float* __restrict__ outF,
               unsigned short* __restrict__ outB,
               int M, int N, int K, int mode, int act) {
  const int tid  = threadIdx.x;
  const int lane = tid & 31;
  const int wid  = tid >> 5;
  const int wm   = wid >> 1;            // 0..3 : 32 rows each
  const int wn   = wid & 1;             // 0..1 : 64 cols each
  const int mBase = blockIdx.y * 128;
  const int nBase = blockIdx.x * 128;

  const int ldr = tid >> 1;             // 0..127 tile row
  const int ldc = (tid & 1) * 16;       // 0 / 16

  v8f zero = {};
  v8f acc[2][4];
#pragma unroll
  for (int i = 0; i < 2; ++i)
#pragma unroll
    for (int j = 0; j < 4; ++j) acc[i][j] = zero;

#if HAVE_ASYNC_LDS
  // Double-buffered LDS, async global->LDS copies overlap WMMA compute.
  __shared__ __align__(16) unsigned short As[2][128 * 32];
  __shared__ __align__(16) unsigned short Bs[2][128 * 32];

  {
    const unsigned short* ag = A + (size_t)(mBase + ldr) * K + ldc;
    const unsigned short* bg = W + (size_t)(nBase + ldr) * K + ldc;
    async_cp16(ag,     &As[0][ldr * 32 + ldc]);
    async_cp16(ag + 8, &As[0][ldr * 32 + ldc + 8]);
    async_cp16(bg,     &Bs[0][ldr * 32 + ldc]);
    async_cp16(bg + 8, &Bs[0][ldr * 32 + ldc + 8]);
  }

  const int KT = K >> 5;
  for (int kt = 0; kt < KT; ++kt) {
    const int cur = kt & 1;
    const int nxt = cur ^ 1;
    WAIT_ASYNC();          // tile kt resident in LDS (this wave's copies)
    __syncthreads();       // all waves' copies visible; buffer nxt free
    if (kt + 1 < KT) {
      const int k0 = (kt + 1) << 5;
      const unsigned short* ag = A + (size_t)(mBase + ldr) * K + k0 + ldc;
      const unsigned short* bg = W + (size_t)(nBase + ldr) * K + k0 + ldc;
      async_cp16(ag,     &As[nxt][ldr * 32 + ldc]);
      async_cp16(ag + 8, &As[nxt][ldr * 32 + ldc + 8]);
      async_cp16(bg,     &Bs[nxt][ldr * 32 + ldc]);
      async_cp16(bg + 8, &Bs[nxt][ldr * 32 + ldc + 8]);
    }
    v16bf af[2];
#pragma unroll
    for (int tm = 0; tm < 2; ++tm)
      af[tm] = load_frag(As[cur], wm * 32 + tm * 16 + (lane & 15), 32, 0, lane);
#pragma unroll
    for (int tn = 0; tn < 4; ++tn) {
      v16bf bf = load_frag(Bs[cur], wn * 64 + tn * 16 + (lane & 15), 32, 0, lane);
#pragma unroll
      for (int tm = 0; tm < 2; ++tm)
        acc[tm][tn] = wmma_bf16(af[tm], bf, acc[tm][tn]);
    }
  }
#else
  // Fallback: synchronous VGPR-staged tile copies.
  __shared__ __align__(16) unsigned short As[128 * 32];
  __shared__ __align__(16) unsigned short Bs[128 * 32];

  for (int k0 = 0; k0 < K; k0 += 32) {
    const unsigned short* ag = A + (size_t)(mBase + ldr) * K + k0 + ldc;
    const unsigned short* bg = W + (size_t)(nBase + ldr) * K + k0 + ldc;
    u32x4 a0 = *(const u32x4*)(ag);
    u32x4 a1 = *(const u32x4*)(ag + 8);
    u32x4 b0 = *(const u32x4*)(bg);
    u32x4 b1 = *(const u32x4*)(bg + 8);
    if (k0 + 32 < K) {                  // hint next K-tile -> global_prefetch_b8
      __builtin_prefetch(ag + 32, 0, 3);
      __builtin_prefetch(bg + 32, 0, 3);
    }
    __syncthreads();
    *(u32x4*)(&As[ldr * 32 + ldc])     = a0;
    *(u32x4*)(&As[ldr * 32 + ldc + 8]) = a1;
    *(u32x4*)(&Bs[ldr * 32 + ldc])     = b0;
    *(u32x4*)(&Bs[ldr * 32 + ldc + 8]) = b1;
    __syncthreads();

    v16bf af[2];
#pragma unroll
    for (int tm = 0; tm < 2; ++tm)
      af[tm] = load_frag(As, wm * 32 + tm * 16 + (lane & 15), 32, 0, lane);
#pragma unroll
    for (int tn = 0; tn < 4; ++tn) {
      v16bf bf = load_frag(Bs, wn * 64 + tn * 16 + (lane & 15), 32, 0, lane);
#pragma unroll
      for (int tm = 0; tm < 2; ++tm)
        acc[tm][tn] = wmma_bf16(af[tm], bf, acc[tm][tn]);
    }
  }
#endif

  // C layout: lanes 0-15 -> N=lane, M=vg; lanes 16-31 -> N=lane-16, M=8+vg
  const int mOff = (lane >> 4) * 8;
  const int nIdx = lane & 15;
#pragma unroll
  for (int tm = 0; tm < 2; ++tm) {
#pragma unroll
    for (int tn = 0; tn < 4; ++tn) {
#pragma unroll
      for (int vg = 0; vg < 8; ++vg) {
        const int m = mBase + wm * 32 + tm * 16 + mOff + vg;
        const int n = nBase + wn * 64 + tn * 16 + nIdx;
        float v = acc[tm][tn][vg];
        if (bias) v += bias[n];
        if (res)  v += res[(size_t)m * N + n];
        if (act == 1) v = v + v / (1.0f + __expf(-v));   // h*sigmoid(h)+h
        if (mode == 0) {
          if (outF) outF[(size_t)m * N + n] = v;
          if (outB) outB[(size_t)m * N + n] = f2bf(v);
        } else if (mode == 1) {
          const int b = m >> 12, s = m & (Sc - 1);
          const int h = n & (Hc - 1), dh = n >> 4;
          outB[(((size_t)(b * Hc + h) * DHc + dh) << 12) + s] = f2bf(v);
        } else {
          const int b = m >> 12, s = m & (Sc - 1);
          const int h = n & (Hc - 1), e = n >> 4;
          outB[((size_t)(b * Hc + h) * Sc + s) * DHc + e] = f2bf(v);
        }
      }
    }
  }
}

// ---------------- attention scores + softmax ----------------
// qp/kp: [B*H][64][4096] bf16 (sequence contiguous = WMMA K axis)
// att:   [B*H][64][64]   bf16 softmax(q@k / 8) rows over e
__global__ __launch_bounds__(128)
void attn_scores(const unsigned short* __restrict__ qp,
                 const unsigned short* __restrict__ kp,
                 unsigned short* __restrict__ att) {
  __shared__ float sc[64][64];
  const int bh   = blockIdx.x;
  const int tid  = threadIdx.x;
  const int lane = tid & 31;
  const int w    = tid >> 5;
  const unsigned short* qb = qp + (size_t)bh * DHc * Sc;
  const unsigned short* kb = kp + (size_t)bh * DHc * Sc;

  v8f zero = {};
  v8f acc[4];
#pragma unroll
  for (int i = 0; i < 4; ++i) acc[i] = zero;

  const int dRow = w * 16 + (lane & 15);
  for (int k0 = 0; k0 < Sc; k0 += 32) {
    v16bf af = load_frag(qb, dRow, Sc, k0, lane);
#pragma unroll
    for (int tn = 0; tn < 4; ++tn) {
      v16bf bf = load_frag(kb, tn * 16 + (lane & 15), Sc, k0, lane);
      acc[tn] = wmma_bf16(af, bf, acc[tn]);
    }
  }
  const int mOff = (lane >> 4) * 8;
  const int nIdx = lane & 15;
#pragma unroll
  for (int tn = 0; tn < 4; ++tn)
#pragma unroll
    for (int vg = 0; vg < 8; ++vg)
      sc[w * 16 + mOff + vg][tn * 16 + nIdx] = acc[tn][vg];
  __syncthreads();

  if (tid < 64) {
    float mx = -3.4e38f;
    for (int e = 0; e < 64; ++e) mx = fmaxf(mx, sc[tid][e] * 0.125f);
    float sum = 0.f;
    for (int e = 0; e < 64; ++e) {
      float xv = __expf(sc[tid][e] * 0.125f - mx);
      sc[tid][e] = xv;
      sum += xv;
    }
    const float inv = 1.0f / sum;
    for (int e = 0; e < 64; ++e)
      att[(size_t)bh * 4096 + tid * 64 + e] = f2bf(sc[tid][e] * inv);
  }
}

// ---------------- attention apply + scrambled reshape ----------------
// a[b,h,d,s] = sum_e att[d,e]*v[e,s]; stored as the reference's
// a_flat[b, d*64 + h*4 + (s>>10), s&1023] (bf16).
__global__ __launch_bounds__(128)
void attn_apply(const unsigned short* __restrict__ att,
                const unsigned short* __restrict__ vt,   // [B*H][S][64]
                unsigned short* __restrict__ aflat) {
  const int bh    = blockIdx.y;
  const int b     = bh >> 4, h = bh & 15;
  const int sBase = blockIdx.x * 128;
  const int tid   = threadIdx.x;
  const int lane  = tid & 31;
  const int w     = tid >> 5;
  const unsigned short* ab = att + (size_t)bh * DHc * DHc;
  const unsigned short* vb = vt  + (size_t)bh * Sc * DHc;

  v8f zero = {};
  v8f acc[8];
#pragma unroll
  for (int i = 0; i < 8; ++i) acc[i] = zero;

  const int dRow = w * 16 + (lane & 15);
#pragma unroll
  for (int k0 = 0; k0 < 64; k0 += 32) {
    v16bf af = load_frag(ab, dRow, 64, k0, lane);
#pragma unroll
    for (int tn = 0; tn < 8; ++tn) {
      v16bf bf = load_frag(vb, sBase + tn * 16 + (lane & 15), 64, k0, lane);
      acc[tn] = wmma_bf16(af, bf, acc[tn]);
    }
  }
  const int mOff = (lane >> 4) * 8;
  const int nIdx = lane & 15;
#pragma unroll
  for (int tn = 0; tn < 8; ++tn) {
#pragma unroll
    for (int vg = 0; vg < 8; ++vg) {
      const int d = w * 16 + mOff + vg;
      const int s = sBase + tn * 16 + nIdx;
      const int r = d * 64 + h * 4 + (s >> 10);
      const int c = s & 1023;
      aflat[((size_t)b * Sc + r) * Dc + c] = f2bf(acc[tn][vg]);
    }
  }
}

// ---------------- driver ----------------
extern "C" void kernel_launch(void* const* d_in, const int* in_sizes, int n_in,
                              void* d_out, int out_size, void* d_ws, size_t ws_size,
                              hipStream_t stream) {
  (void)in_sizes; (void)n_in; (void)out_size; (void)ws_size;

  const float* x   = (const float*)d_in[0];
  const float* scl = (const float*)d_in[1];
  const float* qw  = (const float*)d_in[2];
  const float* qb  = (const float*)d_in[3];
  const float* kw  = (const float*)d_in[4];
  const float* kb  = (const float*)d_in[5];
  const float* vw  = (const float*)d_in[6];
  const float* vb  = (const float*)d_in[7];
  const float* ow  = (const float*)d_in[8];
  const float* ob  = (const float*)d_in[9];
  const float* f1w = (const float*)d_in[10];
  const float* f1b = (const float*)d_in[11];
  const float* f2w = (const float*)d_in[12];
  const float* f2b = (const float*)d_in[13];

  // workspace layout (bf16 elements unless noted)
  unsigned short* wq  = (unsigned short*)d_ws;
  unsigned short* wk  = wq  + (size_t)Dc * Dc;
  unsigned short* wv  = wk  + (size_t)Dc * Dc;
  unsigned short* wo  = wv  + (size_t)Dc * Dc;
  unsigned short* wf1 = wo  + (size_t)Dc * Dc;
  unsigned short* wf2 = wf1 + (size_t)FFc * Dc;
  unsigned short* xnb = wf2 + (size_t)Dc * FFc;            // [16384,1024] (reused for 2nd norm)
  unsigned short* qp  = xnb + (size_t)Mrows * Dc;          // [B,H,DH,S]
  unsigned short* kp  = qp  + (size_t)Mrows * Dc;          // [B,H,DH,S]
  unsigned short* vt  = kp  + (size_t)Mrows * Dc;          // [B,H,S,DH]
  unsigned short* attb  = vt + (size_t)Mrows * Dc;         // [B,H,64,64]
  unsigned short* aflat = attb + (size_t)Bc * Hc * DHc * DHc; // [16384,1024]
  float* xmid = (float*)(aflat + (size_t)Mrows * Dc);      // fp32 [16384,1024]
  unsigned short* hbf = qp;                                // FFN hidden aliases dead qp+kp (64MB)

  // 1) weights -> bf16
  const int wN = Dc * Dc;
  cvt_bf16<<<(wN + 255) / 256, 256, 0, stream>>>(qw,  wq,  wN);
  cvt_bf16<<<(wN + 255) / 256, 256, 0, stream>>>(kw,  wk,  wN);
  cvt_bf16<<<(wN + 255) / 256, 256, 0, stream>>>(vw,  wv,  wN);
  cvt_bf16<<<(wN + 255) / 256, 256, 0, stream>>>(ow,  wo,  wN);
  const int fN = FFc * Dc;
  cvt_bf16<<<(fN + 255) / 256, 256, 0, stream>>>(f1w, wf1, fN);
  cvt_bf16<<<(fN + 255) / 256, 256, 0, stream>>>(f2w, wf2, fN);

  // 2) xn = rms(x) -> bf16
  rmsnorm_bf16<<<Mrows, 256, 0, stream>>>(x, scl, xnb);

  // 3) Q/K/V projections with attention-layout scatter stores
  dim3 g8(Dc / 128, Mrows / 128);
  gemm_bf16<<<g8, 256, 0, stream>>>(xnb, wq, qb, nullptr, nullptr, qp,
                                    Mrows, Dc, Dc, /*mode=*/1, /*act=*/0);
  gemm_bf16<<<g8, 256, 0, stream>>>(xnb, wk, kb, nullptr, nullptr, kp,
                                    Mrows, Dc, Dc, 1, 0);
  gemm_bf16<<<g8, 256, 0, stream>>>(xnb, wv, vb, nullptr, nullptr, vt,
                                    Mrows, Dc, Dc, 2, 0);

  // 4) scores + softmax, 5) apply + scrambled reshape
  attn_scores<<<Bc * Hc, 128, 0, stream>>>(qp, kp, attb);
  attn_apply<<<dim3(Sc / 128, Bc * Hc), 128, 0, stream>>>(attb, vt, aflat);

  // 6) output projection + residual -> xmid (fp32)
  gemm_bf16<<<g8, 256, 0, stream>>>(aflat, wo, ob, x, xmid, nullptr,
                                    Mrows, Dc, Dc, 0, 0);

  // 7) second RMSNorm
  rmsnorm_bf16<<<Mrows, 256, 0, stream>>>(xmid, scl, xnb);

  // 8) FFN up + (h*sigmoid(h)+h) -> bf16
  dim3 g16(FFc / 128, Mrows / 128);
  gemm_bf16<<<g16, 256, 0, stream>>>(xnb, wf1, f1b, nullptr, nullptr, hbf,
                                     Mrows, FFc, Dc, 0, /*act=*/1);

  // 9) FFN down + bias + residual -> d_out (fp32)
  gemm_bf16<<<g8, 256, 0, stream>>>(hbf, wf2, f2b, xmid, (float*)d_out, nullptr,
                                    Mrows, Dc, FFc, 0, 0);
}
